// PC_RNN_GC_14413910245617
// MI455X (gfx1250) — matrix operand order repaired and checked by reading, not compiled, not run on hardware
//
#include <hip/hip_runtime.h>

typedef __bf16 bf16_t;
typedef __attribute__((ext_vector_type(16))) __bf16 v16bf;
typedef __attribute__((ext_vector_type(8)))  __bf16 v8bf;
typedef __attribute__((ext_vector_type(8)))  float   v8f;

#define SEQ_LEN   512
#define BATCH     256
#define OUT_DIM   256
#define STATES    1024
#define TAU_INV   0.1f
#define A_HP      0.1f
#define A_X       0.1f

#define NTHREADS  512
#define NWAVES    16

// padded LDS row strides (bank-conflict mitigation for column-wise epilogue writes)
#define LDF 1032   // f32 rows (16 rows x 1032 floats)
#define LDB 1040   // bf16 activation rows
#define LDE 272    // bf16 error rows (OUT_DIM + pad)

// Hardware tanh (gfx1250 v_tanh_f32). Guarded: builtin if present, else inline asm.
__device__ __forceinline__ float fast_tanh(float v) {
#if __has_builtin(__builtin_amdgcn_tanhf)
  return __builtin_amdgcn_tanhf(v);
#elif __has_builtin(__builtin_amdgcn_tanh_f32)
  return __builtin_amdgcn_tanh_f32(v);
#else
  float r;
  asm("v_tanh_f32 %0, %1" : "=v"(r) : "v"(v));
  return r;
#endif
}

// ---------------------------------------------------------------------------
// Pre-pass: convert an f32 weight matrix into bf16 WMMA-B fragment-major blocks.
// Block (nt, kt) covers B[kt*32 .. kt*32+31][nt*16 .. nt*16+15]; within a block,
// lane l owns 16 contiguous bf16: rows kt*32 + 8*(l>>4) + (e&7) + 16*(e>>3), col nt*16 + (l&15).
// Blocks are ordered k-innermost so the GEMM K-loop streams contiguous 1KB blocks.
// ---------------------------------------------------------------------------
__global__ void pack_bf16_frag(const float* __restrict__ src, bf16_t* __restrict__ dst,
                               int total, int numKt, int trans, int ld) {
  int idx = blockIdx.x * blockDim.x + threadIdx.x;
  if (idx >= total) return;
  int e   = idx & 15;
  int l   = (idx >> 4) & 31;
  int blk = idx >> 9;
  int kt  = blk % numKt;
  int nt  = blk / numKt;
  int row = kt * 32 + ((l >> 4) << 3) + (e & 7) + ((e >> 3) << 4);
  int col = nt * 16 + (l & 15);
  float v = trans ? src[(size_t)col * ld + row] : src[(size_t)row * ld + col];
  dst[idx] = (bf16_t)v;
}

// ---------------------------------------------------------------------------
// Dual-tile GEMM: two 16x16 f32 output tiles sharing one A stream.
// Software-pipelined: A(LDS)/B(L2) fragments for k+1 are issued before the
// WMMAs of k, so s_wait_* overlaps with the tensor pipe.
// ---------------------------------------------------------------------------
__device__ __forceinline__ void gemm_tile2(const bf16_t* __restrict__ sA, int ldsA,
                                           const bf16_t* __restrict__ wB,
                                           int nt0, int nt1, int numKt, int lane,
                                           v8f& acc0, v8f& acc1) {
  const int m    = lane & 15;
  const int khlf = (lane >> 4) * 8;            // A layout: hi lanes own K 8..15 / 24..31
  const bf16_t* __restrict__ aRow = sA + m * ldsA;
  const bf16_t* __restrict__ b0 = wB + ((size_t)nt0 * numKt) * 512 + lane * 16;
  const bf16_t* __restrict__ b1 = wB + ((size_t)nt1 * numKt) * 512 + lane * 16;

  // prologue: stage k = 0
  v8bf  aL  = *(const v8bf*)(aRow + khlf);
  v8bf  aH  = *(const v8bf*)(aRow + khlf + 16);
  v16bf vb0 = *(const v16bf*)(b0);
  v16bf vb1 = *(const v16bf*)(b1);

#pragma unroll
  for (int kt = 0; kt < numKt; ++kt) {
    v16bf a;
#pragma unroll
    for (int i = 0; i < 8; ++i) { a[i] = aL[i]; a[i + 8] = aH[i]; }
    v16bf cb0 = vb0, cb1 = vb1;
    if (kt + 1 < numKt) {                      // preload k+1 before consuming k
      const int k1 = (kt + 1) * 32 + khlf;
      aL  = *(const v8bf*)(aRow + k1);
      aH  = *(const v8bf*)(aRow + k1 + 16);
      vb0 = *(const v16bf*)(b0 + (size_t)(kt + 1) * 512);
      vb1 = *(const v16bf*)(b1 + (size_t)(kt + 1) * 512);
      if (((kt + 1) & 3) == 0) {               // prefetch 8 blocks ahead
        __builtin_prefetch(b0 + (size_t)(kt + 8) * 512, 0, 3);
        __builtin_prefetch(b1 + (size_t)(kt + 8) * 512, 0, 3);
      }
    }
    acc0 = __builtin_amdgcn_wmma_f32_16x16x32_bf16(false, a, false, cb0,
                                                   (short)0, acc0, false, false);
    acc1 = __builtin_amdgcn_wmma_f32_16x16x32_bf16(false, a, false, cb1,
                                                   (short)0, acc1, false, false);
  }
}

// Single-tile pipelined variant (used when only one tile per wave: P4).
__device__ __forceinline__ v8f gemm_tile(const bf16_t* __restrict__ sA, int ldsA,
                                         const bf16_t* __restrict__ wB,
                                         int nt, int numKt, int lane) {
  v8f acc = {};
  const int m    = lane & 15;
  const int khlf = (lane >> 4) * 8;
  const bf16_t* __restrict__ aRow = sA + m * ldsA;
  const bf16_t* __restrict__ bp = wB + ((size_t)nt * numKt) * 512 + lane * 16;

  v8bf  aL = *(const v8bf*)(aRow + khlf);
  v8bf  aH = *(const v8bf*)(aRow + khlf + 16);
  v16bf vb = *(const v16bf*)(bp);

#pragma unroll
  for (int kt = 0; kt < numKt; ++kt) {
    v16bf a;
#pragma unroll
    for (int i = 0; i < 8; ++i) { a[i] = aL[i]; a[i + 8] = aH[i]; }
    v16bf cb = vb;
    if (kt + 1 < numKt) {
      const int k1 = (kt + 1) * 32 + khlf;
      aL = *(const v8bf*)(aRow + k1);
      aH = *(const v8bf*)(aRow + k1 + 16);
      vb = *(const v16bf*)(bp + (size_t)(kt + 1) * 512);
      if (((kt + 1) & 3) == 0)
        __builtin_prefetch(bp + (size_t)(kt + 8) * 512, 0, 3);
    }
    acc = __builtin_amdgcn_wmma_f32_16x16x32_bf16(false, a, false, cb,
                                                  (short)0, acc, false, false);
  }
  return acc;
}

// ---------------------------------------------------------------------------
// Persistent PC-RNN kernel: 1 workgroup = 16 batch rows for all 512 steps.
// All state lives in LDS; weights stream from L2 in bf16 fragment blocks.
// Per step: 4 barrier-separated phases; th/thp bf16 copies are cached in LDS
// so each state element needs only 2 hardware-tanh evaluations per step.
// ---------------------------------------------------------------------------
__global__ __launch_bounds__(NTHREADS)
void pc_rnn_kernel(const float* __restrict__ x,
                   const float* __restrict__ h_init,
                   const float* __restrict__ b_o,
                   const float* __restrict__ b_r,
                   const bf16_t* __restrict__ wrT,   // B = w_r^T  (K=1024, N=1024)
                   const bf16_t* __restrict__ wr,    // B = w_r    (K=1024, N=1024)
                   const bf16_t* __restrict__ woT,   // B = w_o^T  (K=1024, N=256)
                   const bf16_t* __restrict__ wo,    // B = w_o    (K=256,  N=1024)
                   float* __restrict__ out) {
  extern __shared__ char smem_raw[];
  float*  sh_post  = (float*)smem_raw;                 // 16 x LDF  f32 (persistent)
  float*  sh_prime = sh_post  + 16 * LDF;              // 16 x LDF  f32 (persistent)
  float*  s_tmp    = sh_prime + 16 * LDF;              // 16 x LDF  f32 (ehp -> h_prior)
  bf16_t* s_abf    = (bf16_t*)(s_tmp + 16 * LDF);      // 16 x LDB  bf16 (th <-> thp)
  bf16_t* s_bbf    = s_abf + 16 * LDB;                 // 16 x LDB  bf16 (ehp)
  bf16_t* s_ebf    = s_bbf + 16 * LDB;                 // 16 x LDE  bf16 (error)

  const int tid    = threadIdx.x;
  const int wave   = tid >> 5;
  const int lane   = tid & 31;
  const int batch0 = blockIdx.x * 16;
  const int colq   = lane & 15;                // column within tile
  const int m0     = (lane >> 4) * 8;          // first accum row for this lane

  // Init persistent state: h_post from h_init, h_prime = 0, th = tanh(h_post).
  for (int i = tid; i < 16 * STATES; i += NTHREADS) {
    int m = i >> 10, k = i & (STATES - 1);
    float h = h_init[(size_t)(batch0 + m) * STATES + k];
    sh_post [m * LDF + k] = h;
    sh_prime[m * LDF + k] = 0.0f;
    s_abf   [m * LDB + k] = (bf16_t)fast_tanh(h);
  }
  __syncthreads();

#pragma unroll 1
  for (int t = 0; t < SEQ_LEN; ++t) {
    // ---- P2: r1 = th @ w_r^T ; ehp / h_prime update fused in epilogue ----
    {
      auto ep = [&](int nt, const v8f& acc) {
        const int col = nt * 16 + colq;
        const float br = b_r[col];
#pragma unroll
        for (int r = 0; r < 8; ++r) {
          const int m = m0 + r;
          float hp    = sh_prime[m * LDF + col];
          float hpost = sh_post [m * LDF + col];
          float ehp   = hp - (acc[r] + br - hpost) * TAU_INV;
          sh_prime[m * LDF + col] = hp - A_HP * ehp;
          s_tmp   [m * LDF + col] = ehp;
          s_bbf   [m * LDB + col] = (bf16_t)ehp;
        }
      };
#pragma unroll 1
      for (int p = 0; p < 2; ++p) {
        const int nt0 = wave + (2 * p) * NWAVES;
        const int nt1 = nt0 + NWAVES;
        v8f acc0 = {}, acc1 = {};
        gemm_tile2(s_abf, LDB, wrT, nt0, nt1, STATES / 32, lane, acc0, acc1);
        ep(nt0, acc0);
        ep(nt1, acc1);
      }
    }
    __syncthreads();

    // ---- P3: r2 = ehp @ w_r ; h_prior fused; th read back from LDS; thp cached --
    {
      auto ep = [&](int nt, const v8f& acc) {
        const int col = nt * 16 + colq;
#pragma unroll
        for (int r = 0; r < 8; ++r) {
          const int m = m0 + r;
          float ehp    = s_tmp  [m * LDF + col];
          float hpost  = sh_post[m * LDF + col];
          float th     = (float)s_abf[m * LDB + col];   // cached tanh(h_post)
          float hprior = hpost + sh_prime[m * LDF + col]
                       + (A_HP * TAU_INV) * ((1.0f - th * th) * acc[r] - ehp);
          s_tmp[m * LDF + col] = hprior;
          s_abf[m * LDB + col] = (bf16_t)fast_tanh(hprior);  // thp
        }
      };
#pragma unroll 1
      for (int p = 0; p < 2; ++p) {
        const int nt0 = wave + (2 * p) * NWAVES;
        const int nt1 = nt0 + NWAVES;
        v8f acc0 = {}, acc1 = {};
        gemm_tile2(s_bbf, LDB, wr, nt0, nt1, STATES / 32, lane, acc0, acc1);
        ep(nt0, acc0);
        ep(nt1, acc1);
      }
    }
    __syncthreads();

    // ---- P4: x_pred = thp @ w_o^T + b_o ; error -> global out + LDS bf16 ----
    {
      const int nt  = wave;                       // OUT_DIM/16 == NWAVES tiles
      v8f acc = gemm_tile(s_abf, LDB, woT, nt, STATES / 32, lane);
      const int col = nt * 16 + colq;
      const float bo = b_o[col];
      const size_t xbase = (size_t)t * (BATCH * OUT_DIM) + (size_t)batch0 * OUT_DIM;
#pragma unroll
      for (int r = 0; r < 8; ++r) {
        const int m = m0 + r;
        float err = acc[r] + bo - x[xbase + (size_t)m * OUT_DIM + col];
        out[xbase + (size_t)m * OUT_DIM + col] = err;
        s_ebf[m * LDE + col] = (bf16_t)err;
      }
    }
    __syncthreads();

    // ---- P5: h_post = h_prior - a_x*(1-thp^2)*(err @ w_o) ; th for next step ----
    {
      auto ep = [&](int nt, const v8f& acc) {
        const int col = nt * 16 + colq;
#pragma unroll
        for (int r = 0; r < 8; ++r) {
          const int m = m0 + r;
          float hprior = s_tmp[m * LDF + col];
          float thp    = (float)s_abf[m * LDB + col];   // cached tanh(h_prior)
          float hpost  = hprior - A_X * (1.0f - thp * thp) * acc[r];
          sh_post[m * LDF + col] = hpost;
          s_abf  [m * LDB + col] = (bf16_t)fast_tanh(hpost);  // th for next step
        }
      };
#pragma unroll 1
      for (int p = 0; p < 2; ++p) {
        const int nt0 = wave + (2 * p) * NWAVES;
        const int nt1 = nt0 + NWAVES;
        v8f acc0 = {}, acc1 = {};
        gemm_tile2(s_ebf, LDE, wo, nt0, nt1, OUT_DIM / 32, lane, acc0, acc1);
        ep(nt0, acc0);
        ep(nt1, acc1);
      }
    }
    __syncthreads();
  }
}

// ---------------------------------------------------------------------------
extern "C" void kernel_launch(void* const* d_in, const int* in_sizes, int n_in,
                              void* d_out, int out_size, void* d_ws, size_t ws_size,
                              hipStream_t stream) {
  const float* x      = (const float*)d_in[0];
  const float* h_init = (const float*)d_in[1];
  const float* w_o    = (const float*)d_in[2];
  const float* b_o    = (const float*)d_in[3];
  const float* w_r    = (const float*)d_in[4];
  const float* b_r    = (const float*)d_in[5];
  float* out = (float*)d_out;

  // ws layout: 4 fragment-packed bf16 weight images (5 MB total, L2-resident)
  bf16_t* wrT = (bf16_t*)d_ws;                  // 1024x1024
  bf16_t* wr  = wrT + (size_t)STATES * STATES;  // 1024x1024
  bf16_t* woT = wr  + (size_t)STATES * STATES;  // K=1024, N=256
  bf16_t* wo  = woT + (size_t)STATES * OUT_DIM; // K=256,  N=1024

  const int totRR = STATES * STATES;   // 1M elements
  const int totRO = STATES * OUT_DIM;  // 256K elements
  // B = w_r^T : B(k,j) = w_r[j][k]
  pack_bf16_frag<<<(totRR + 255) / 256, 256, 0, stream>>>(w_r, wrT, totRR, STATES / 32, 1, STATES);
  // B = w_r   : B(k,j) = w_r[k][j]
  pack_bf16_frag<<<(totRR + 255) / 256, 256, 0, stream>>>(w_r, wr,  totRR, STATES / 32, 0, STATES);
  // B = w_o^T : K=1024, N=256, B(k,j) = w_o[j][k]
  pack_bf16_frag<<<(totRO + 255) / 256, 256, 0, stream>>>(w_o, woT, totRO, STATES / 32, 1, STATES);
  // B = w_o   : K=256, N=1024, B(k,j) = w_o[k][j]
  pack_bf16_frag<<<(totRO + 255) / 256, 256, 0, stream>>>(w_o, wo,  totRO, OUT_DIM / 32, 0, STATES);

  const size_t smem = (3 * 16 * LDF) * sizeof(float)
                    + (2 * 16 * LDB + 16 * LDE) * sizeof(bf16_t);  // ~273 KB
  pc_rnn_kernel<<<BATCH / 16, NTHREADS, smem, stream>>>(
      x, h_init, b_o, b_r, wrT, wr, woT, wo, out);
}
